// SelfAttention_50036368998914
// MI455X (gfx1250) — matrix operand order repaired and checked by reading, not compiled, not run on hardware
//
#include <hip/hip_runtime.h>
#include <hip/hip_bf16.h>

// ---------------------------------------------------------------------------
// CDNA5 (gfx1250) bf16 WMMA self-attention.
// All matmuls are expressed as NT GEMM: C[m,n] = sum_k A[m,k] * Bt[n,k]
// so both A and B fragments of v_wmma_f32_16x16x32_bf16 load from contiguous
// rows (A: two 16B chunks / lane, B: two 16B chunks / lane).
// ---------------------------------------------------------------------------

typedef __bf16 bf16_t;
typedef bf16_t v16bf __attribute__((ext_vector_type(16)));
typedef bf16_t v8bf  __attribute__((ext_vector_type(8)));
typedef float  v8f   __attribute__((ext_vector_type(8)));
typedef float  v4f   __attribute__((ext_vector_type(4)));
typedef int    v4i   __attribute__((ext_vector_type(4)));

#define EMB   1024
#define SEQ   2048
#define BATCH 4
#define MTOT  (BATCH * SEQ)   // 8192 rows total

// ---------------- fp32 -> bf16 conversion (8 elements / thread) ------------
__global__ __launch_bounds__(256)
void cvt_f32_to_bf16(const float* __restrict__ in, bf16_t* __restrict__ out, int n)
{
    int i = (blockIdx.x * 256 + threadIdx.x) * 8;
    if (i >= n) return;
    v4f a = *(const v4f*)(in + i);
    v4f b = *(const v4f*)(in + i + 4);
    v8bf o;
    o[0] = (bf16_t)a[0]; o[1] = (bf16_t)a[1]; o[2] = (bf16_t)a[2]; o[3] = (bf16_t)a[3];
    o[4] = (bf16_t)b[0]; o[5] = (bf16_t)b[1]; o[6] = (bf16_t)b[2]; o[7] = (bf16_t)b[3];
    *(v8bf*)(out + i) = o;
}

// ---------------- WMMA fragment loaders (CDNA5 wave32 layouts) -------------
// A (16x32, MxK, bf16): lane = g*16 + r holds row M=r;
//   VGPR0..3 = K[8g .. 8g+7], VGPR4..7 = K[16+8g .. 16+8g+7]
__device__ __forceinline__
v16bf load_a_frag(const bf16_t* __restrict__ A, int lda, int m0, int k0, int lane)
{
    int g = lane >> 4, r = lane & 15;
    const bf16_t* p = A + (size_t)(m0 + r) * lda + (k0 + 8 * g);
    v8bf lo = *(const v8bf*)p;         // K = 8g .. 8g+7
    v8bf hi = *(const v8bf*)(p + 16);  // K = 16+8g .. 16+8g+7
    return __builtin_shufflevector(lo, hi, 0,1,2,3,4,5,6,7,8,9,10,11,12,13,14,15);
}

// B (32x16, KxN, bf16) supplied as Bt row-major (N x K): lane = g*16 + c holds
// column N=c; VGPR0..3 = K[8g .. 8g+7], VGPR4..7 = K[16+8g .. 16+8g+7].
__device__ __forceinline__
v16bf load_b_frag(const bf16_t* __restrict__ Bt, int ldb, int n0, int k0, int lane)
{
    int g = lane >> 4, c = lane & 15;
    const bf16_t* p = Bt + (size_t)(n0 + c) * ldb + (k0 + 8 * g);
    v8bf lo = *(const v8bf*)p;
    v8bf hi = *(const v8bf*)(p + 16);
    return __builtin_shufflevector(lo, hi, 0,1,2,3,4,5,6,7,8,9,10,11,12,13,14,15);
}

// ---------------- NT GEMM: 128x128 block tile, 8 waves, 32x64 / wave -------
enum { OUT_BF16 = 0, OUT_F32_SCALED = 1, OUT_F32_BIAS = 2 };

template <int MODE>
__global__ __launch_bounds__(256)
void gemm_nt(const bf16_t* __restrict__ A, const bf16_t* __restrict__ Bt,
             void* __restrict__ Cvoid, const float* __restrict__ bias, float alpha,
             int K, int lda, int ldb, int ldc,
             long long aStride, long long bStride, long long cStride)
{
    A  += (long long)blockIdx.z * aStride;
    Bt += (long long)blockIdx.z * bStride;

    const int tid  = threadIdx.x;
    const int wave = tid >> 5;
    const int lane = tid & 31;
    const int wm   = wave >> 1;   // 0..3
    const int wn   = wave & 1;    // 0..1
    const int m0   = blockIdx.y * 128 + wm * 32;   // wave: 32 rows
    const int n0   = blockIdx.x * 128 + wn * 64;   // wave: 64 cols

    v8f acc[2][4] = {};

    for (int k0 = 0; k0 < K; k0 += 32) {
        v16bf a0 = load_a_frag(A, lda, m0,      k0, lane);
        v16bf a1 = load_a_frag(A, lda, m0 + 16, k0, lane);
        v16bf b0 = load_b_frag(Bt, ldb, n0,      k0, lane);
        v16bf b1 = load_b_frag(Bt, ldb, n0 + 16, k0, lane);
        v16bf b2 = load_b_frag(Bt, ldb, n0 + 32, k0, lane);
        v16bf b3 = load_b_frag(Bt, ldb, n0 + 48, k0, lane);

        acc[0][0] = __builtin_amdgcn_wmma_f32_16x16x32_bf16(false, a0, false, b0, (short)0, acc[0][0], false, false);
        acc[0][1] = __builtin_amdgcn_wmma_f32_16x16x32_bf16(false, a0, false, b1, (short)0, acc[0][1], false, false);
        acc[0][2] = __builtin_amdgcn_wmma_f32_16x16x32_bf16(false, a0, false, b2, (short)0, acc[0][2], false, false);
        acc[0][3] = __builtin_amdgcn_wmma_f32_16x16x32_bf16(false, a0, false, b3, (short)0, acc[0][3], false, false);
        acc[1][0] = __builtin_amdgcn_wmma_f32_16x16x32_bf16(false, a1, false, b0, (short)0, acc[1][0], false, false);
        acc[1][1] = __builtin_amdgcn_wmma_f32_16x16x32_bf16(false, a1, false, b1, (short)0, acc[1][1], false, false);
        acc[1][2] = __builtin_amdgcn_wmma_f32_16x16x32_bf16(false, a1, false, b2, (short)0, acc[1][2], false, false);
        acc[1][3] = __builtin_amdgcn_wmma_f32_16x16x32_bf16(false, a1, false, b3, (short)0, acc[1][3], false, false);
    }

    // D layout (f32 16x16): lane = g*16 + c holds col N=c, VGPRv -> row M = 8g+v
    const int g = lane >> 4, c = lane & 15;
    #pragma unroll
    for (int mi = 0; mi < 2; ++mi) {
        #pragma unroll
        for (int ni = 0; ni < 4; ++ni) {
            const int row0 = m0 + mi * 16 + g * 8;
            const int col  = n0 + ni * 16 + c;
            #pragma unroll
            for (int v = 0; v < 8; ++v) {
                const size_t idx = (size_t)(row0 + v) * ldc + col;
                if (MODE == OUT_BF16) {
                    bf16_t* C = (bf16_t*)Cvoid + (long long)blockIdx.z * cStride;
                    C[idx] = (bf16_t)acc[mi][ni][v];
                } else if (MODE == OUT_F32_SCALED) {
                    float* C = (float*)Cvoid + (long long)blockIdx.z * cStride;
                    C[idx] = acc[mi][ni][v] * alpha;
                } else {
                    float* C = (float*)Cvoid + (long long)blockIdx.z * cStride;
                    C[idx] = acc[mi][ni][v] + bias[col];
                }
            }
        }
    }
}

// ---------------- masked softmax: one 256-thread block per score row -------
__global__ __launch_bounds__(256)
void masked_softmax(const float* __restrict__ scores, const int* __restrict__ mask,
                    bf16_t* __restrict__ attn)
{
    const long long row = blockIdx.x;               // b*SEQ + q
    const float* sr = scores + row * SEQ;
    const int*   mr = mask   + row * SEQ;
    bf16_t*      ar = attn   + row * SEQ;

    const int t = threadIdx.x;
    float v[8];
    float mx = -3.0e38f;
    {
        const int i = t * 8;
        v4f s0 = *(const v4f*)(sr + i);
        v4f s1 = *(const v4f*)(sr + i + 4);
        v4i m0 = *(const v4i*)(mr + i);
        v4i m1 = *(const v4i*)(mr + i + 4);
        #pragma unroll
        for (int j = 0; j < 4; ++j) {
            v[j]     = (m0[j] == 0) ? -1.0e9f : s0[j];
            v[j + 4] = (m1[j] == 0) ? -1.0e9f : s1[j];
        }
        #pragma unroll
        for (int j = 0; j < 8; ++j) mx = fmaxf(mx, v[j]);
    }

    __shared__ float red[256];
    red[t] = mx; __syncthreads();
    for (int off = 128; off > 0; off >>= 1) {
        if (t < off) red[t] = fmaxf(red[t], red[t + off]);
        __syncthreads();
    }
    mx = red[0];
    __syncthreads();

    float sum = 0.0f;
    #pragma unroll
    for (int j = 0; j < 8; ++j) { v[j] = __expf(v[j] - mx); sum += v[j]; }
    red[t] = sum; __syncthreads();
    for (int off = 128; off > 0; off >>= 1) {
        if (t < off) red[t] += red[t + off];
        __syncthreads();
    }
    const float inv = 1.0f / red[0];

    v8bf o;
    #pragma unroll
    for (int j = 0; j < 8; ++j) o[j] = (bf16_t)(v[j] * inv);
    *(v8bf*)(ar + t * 8) = o;
}

// ---------------------------------------------------------------------------
extern "C" void kernel_launch(void* const* d_in, const int* in_sizes, int n_in,
                              void* d_out, int out_size, void* d_ws, size_t ws_size,
                              hipStream_t stream)
{
    const float* x    = (const float*)d_in[0];
    const int*   mask = (const int*)  d_in[1];
    const float* Wq   = (const float*)d_in[2];
    const float* Wk   = (const float*)d_in[3];
    const float* Wv   = (const float*)d_in[4];
    const float* Wo   = (const float*)d_in[5];
    const float* bo   = (const float*)d_in[6];
    float*       out  = (float*)d_out;

    // ---- workspace carve-up (~184 MB) ----
    char* ws = (char*)d_ws;
    size_t off = 0;
    bf16_t* xb   = (bf16_t*)(ws + off); off += (size_t)MTOT * EMB * 2;         // 16 MB
    bf16_t* wqb  = (bf16_t*)(ws + off); off += (size_t)EMB * EMB * 2;          //  2 MB
    bf16_t* wkb  = (bf16_t*)(ws + off); off += (size_t)EMB * EMB * 2;
    bf16_t* wvb  = (bf16_t*)(ws + off); off += (size_t)EMB * EMB * 2;
    bf16_t* wob  = (bf16_t*)(ws + off); off += (size_t)EMB * EMB * 2;
    bf16_t* Qb   = (bf16_t*)(ws + off); off += (size_t)MTOT * EMB * 2;         // 16 MB
    bf16_t* Kb   = (bf16_t*)(ws + off); off += (size_t)MTOT * EMB * 2;         // 16 MB
    bf16_t* Vtb  = (bf16_t*)(ws + off); off += (size_t)BATCH * EMB * SEQ * 2;  // 16 MB
    float*  scor = (float*) (ws + off); off += (size_t)BATCH * SEQ * SEQ * 4;  // 64 MB
    bf16_t* attn = (bf16_t*)(ws + off); off += (size_t)BATCH * SEQ * SEQ * 2;  // 32 MB
    bf16_t* ao   = (bf16_t*)(ws + off); off += (size_t)MTOT * EMB * 2;         // 16 MB

    // ---- 1) fp32 -> bf16 ----
    {
        int n = MTOT * EMB;
        cvt_f32_to_bf16<<<n / (256 * 8), 256, 0, stream>>>(x, xb, n);
        int nw = EMB * EMB;
        cvt_f32_to_bf16<<<nw / (256 * 8), 256, 0, stream>>>(Wq, wqb, nw);
        cvt_f32_to_bf16<<<nw / (256 * 8), 256, 0, stream>>>(Wk, wkb, nw);
        cvt_f32_to_bf16<<<nw / (256 * 8), 256, 0, stream>>>(Wv, wvb, nw);
        cvt_f32_to_bf16<<<nw / (256 * 8), 256, 0, stream>>>(Wo, wob, nw);
    }

    // ---- 2) Q = x @ Wq^T ; K = x @ Wk^T   (M=8192, N=1024, K=1024) ----
    {
        dim3 grid(EMB / 128, MTOT / 128, 1);
        gemm_nt<OUT_BF16><<<grid, 256, 0, stream>>>(
            xb, wqb, Qb, nullptr, 1.0f, EMB, EMB, EMB, EMB, 0LL, 0LL, 0LL);
        gemm_nt<OUT_BF16><<<grid, 256, 0, stream>>>(
            xb, wkb, Kb, nullptr, 1.0f, EMB, EMB, EMB, EMB, 0LL, 0LL, 0LL);
    }

    // ---- 3) Vt = Wv @ x^T  per batch (M=E=1024, N=S=2048, K=1024) ----
    {
        dim3 grid(SEQ / 128, EMB / 128, BATCH);
        gemm_nt<OUT_BF16><<<grid, 256, 0, stream>>>(
            wvb, xb, Vtb, nullptr, 1.0f, EMB, EMB, EMB, SEQ,
            0LL, (long long)SEQ * EMB, (long long)EMB * SEQ);
    }

    // ---- 4) scores = (Q @ K^T) / sqrt(E)  per batch (M=N=2048, K=1024) ----
    {
        dim3 grid(SEQ / 128, SEQ / 128, BATCH);
        gemm_nt<OUT_F32_SCALED><<<grid, 256, 0, stream>>>(
            Qb, Kb, scor, nullptr, 0.03125f /* 1024^-0.5 */, EMB, EMB, EMB, SEQ,
            (long long)SEQ * EMB, (long long)SEQ * EMB, (long long)SEQ * SEQ);
    }

    // ---- 5) attn = softmax(mask ? scores : -1e9)  -> bf16 ----
    masked_softmax<<<BATCH * SEQ, 256, 0, stream>>>(scor, mask, attn);

    // ---- 6) ao = attn @ V = attn @ (Vt)^T  per batch (M=2048, N=1024, K=2048) ----
    {
        dim3 grid(EMB / 128, SEQ / 128, BATCH);
        gemm_nt<OUT_BF16><<<grid, 256, 0, stream>>>(
            attn, Vtb, ao, nullptr, 1.0f, SEQ, SEQ, SEQ, EMB,
            (long long)SEQ * SEQ, (long long)EMB * SEQ, (long long)SEQ * EMB);
    }

    // ---- 7) out = ao @ Wo^T + bo  (M=8192, N=1024, K=1024, fp32 out) ----
    {
        dim3 grid(EMB / 128, MTOT / 128, 1);
        gemm_nt<OUT_F32_BIAS><<<grid, 256, 0, stream>>>(
            ao, wob, out, bo, 1.0f, EMB, EMB, EMB, EMB, 0LL, 0LL, 0LL);
    }
}
